// FastAttention_70102456205557
// MI455X (gfx1250) — compile-verified, matrix operand order checked by reading, not run on hardware
//
#include <hip/hip_runtime.h>
#include <hip/hip_bf16.h>

typedef __attribute__((ext_vector_type(16))) __bf16 v16bf;
typedef __attribute__((ext_vector_type(8)))  __bf16 v8bf;
typedef __attribute__((ext_vector_type(8)))  float  v8f;

#define DIMK 768
#define NROWS 32768            // B*N
#define KT_STEPS 24            // 768/32
#define CT_TOTAL 192           // 4*768/16 column tiles across Wq|Wk|Wv|Wr
#define SCALE 0.036084391824351615f  // 768^-0.5
#define LA_BUF 4096            // 8 subtiles * 32 lanes * 16 elems (bf16) per stage buffer

// ---------------------------------------------------------------------------
// Kernel 0: pack Wq|Wk|Wv|Wr (f32, row-major K x N) into bf16 B-fragments.
// lane<16  -> col=lane,     K = {k0..k0+7,  k0+16..k0+23}
// lane>=16 -> col=lane-16,  K = {k0+8..k0+15, k0+24..k0+31}
// ---------------------------------------------------------------------------
__global__ void pack_weights(const float* __restrict__ Wq, const float* __restrict__ Wk,
                             const float* __restrict__ Wv, const float* __restrict__ Wr,
                             __bf16* __restrict__ Bp) {
  int gid = blockIdx.x * blockDim.x + threadIdx.x;   // 0 .. 192*24*32-1
  int ct   = gid / (KT_STEPS * 32);
  int rem  = gid % (KT_STEPS * 32);
  int kt   = rem >> 5;
  int lane = rem & 31;
  int ncol   = ct * 16 + (lane & 15);
  int tensor = ncol / DIMK;
  int n      = ncol % DIMK;
  const float* W = (tensor == 0) ? Wq : (tensor == 1) ? Wk : (tensor == 2) ? Wv : Wr;
  int kbase = kt * 32 + ((lane >> 4) * 8);
  __bf16* dst = Bp + (size_t)gid * 16;
#pragma unroll
  for (int e = 0; e < 16; ++e) {
    int k = kbase + ((e < 8) ? e : (8 + e));   // e>=8 -> kbase+16+(e-8)
    dst[e] = (__bf16)W[(size_t)k * DIMK + n];
  }
}

__global__ void zero_f32(float* __restrict__ p, int n) {
  int i = blockIdx.x * blockDim.x + threadIdx.x;
  if (i < n) p[i] = 0.0f;
}

// ---------------------------------------------------------------------------
// Kernel 1: fused q/k/v GEMM.  C[32768 x 2304] = x[32768x768] @ [Wq|Wk|Wv].
// 256 thr (8 waves), tile 128 rows x 128 cols, wave = 2 Msub x 4 Nsub.
// Double-buffered LDS A staging (one barrier per K-step).
// ---------------------------------------------------------------------------
__global__ __launch_bounds__(256) void gemm_qkv(const float* __restrict__ x,
                                                const __bf16* __restrict__ Bp,
                                                float* __restrict__ qo,
                                                __bf16* __restrict__ ko,
                                                __bf16* __restrict__ vo) {
  __shared__ __bf16 lA[2 * LA_BUF];    // 16 KB, double buffered
  int tid  = threadIdx.x;
  int lane = tid & 31;
  int wave = tid >> 5;
  int wm   = wave & 3;                 // M group (2 subtiles each)
  int wn   = wave >> 2;                // N group (4 subtiles each)
  int mbase = blockIdx.x * 128;
  int cbase = blockIdx.y * 128;        // 0..2303, never straddles a tensor
  int tensor = cbase / DIMK;
  int cloc   = cbase - tensor * DIMK;

  v8f acc[2][4];
#pragma unroll
  for (int i = 0; i < 2; ++i)
#pragma unroll
    for (int j = 0; j < 4; ++j)
#pragma unroll
      for (int r = 0; r < 8; ++r) acc[i][j][r] = 0.0f;

  // staging geometry: thread tid fills fragment slot (sub = wave, lane)
  int sm    = mbase + wave * 16 + (lane & 15);
  int skoff = (lane >> 4) * 8;
  const float* srow = x + (size_t)sm * DIMK + skoff;

  auto stage = [&](int kt, int buf) {
    const float* s = srow + kt * 32;
    float4 f0 = *(const float4*)(s);
    float4 f1 = *(const float4*)(s + 4);
    float4 g0 = *(const float4*)(s + 16);
    float4 g1 = *(const float4*)(s + 20);
    __bf16* d = &lA[buf * LA_BUF + tid * 16];
    d[0]=(__bf16)f0.x; d[1]=(__bf16)f0.y; d[2]=(__bf16)f0.z; d[3]=(__bf16)f0.w;
    d[4]=(__bf16)f1.x; d[5]=(__bf16)f1.y; d[6]=(__bf16)f1.z; d[7]=(__bf16)f1.w;
    d[8]=(__bf16)g0.x; d[9]=(__bf16)g0.y; d[10]=(__bf16)g0.z; d[11]=(__bf16)g0.w;
    d[12]=(__bf16)g1.x; d[13]=(__bf16)g1.y; d[14]=(__bf16)g1.z; d[15]=(__bf16)g1.w;
  };

  stage(0, 0);
  for (int kt = 0; kt < KT_STEPS; ++kt) {
    __syncthreads();
    int cur = kt & 1;
    if (kt + 1 < KT_STEPS) {
      __builtin_prefetch(srow + (kt + 2) * 32, 0, 3);  // global_prefetch_b8
      stage(kt + 1, cur ^ 1);
    }
    v16bf a0 = *(const v16bf*)&lA[cur * LA_BUF + ((2 * wm + 0) * 32 + lane) * 16];
    v16bf a1 = *(const v16bf*)&lA[cur * LA_BUF + ((2 * wm + 1) * 32 + lane) * 16];
#pragma unroll
    for (int j = 0; j < 4; ++j) {
      int ct = (cbase >> 4) + wn * 4 + j;
      v16bf bfrag = *(const v16bf*)(Bp + (((size_t)ct * KT_STEPS + kt) * 32 + lane) * 16);
      acc[0][j] = __builtin_amdgcn_wmma_f32_16x16x32_bf16(false, a0, false, bfrag,
                                                          (short)0, acc[0][j], false, false);
      acc[1][j] = __builtin_amdgcn_wmma_f32_16x16x32_bf16(false, a1, false, bfrag,
                                                          (short)0, acc[1][j], false, false);
    }
  }

  // epilogue: C/D layout -> lane<16: M=r, lane>=16: M=r+8; N = lane&15.
  // tensor is block-uniform: hoist branch outside the store loops.
  if (tensor == 0) {
#pragma unroll
    for (int i = 0; i < 2; ++i) {
      int rb = mbase + (2 * wm + i) * 16 + ((lane < 16) ? 0 : 8);
#pragma unroll
      for (int j = 0; j < 4; ++j) {
        int c = cloc + (wn * 4 + j) * 16 + (lane & 15);
#pragma unroll
        for (int r = 0; r < 8; ++r)
          qo[(size_t)(rb + r) * DIMK + c] = acc[i][j][r];
      }
    }
  } else {
    __bf16* dst = (tensor == 1) ? ko : vo;
#pragma unroll
    for (int i = 0; i < 2; ++i) {
      int rb = mbase + (2 * wm + i) * 16 + ((lane < 16) ? 0 : 8);
#pragma unroll
      for (int j = 0; j < 4; ++j) {
        int c = cloc + (wn * 4 + j) * 16 + (lane & 15);
#pragma unroll
        for (int r = 0; r < 8; ++r)
          dst[(size_t)(rb + r) * DIMK + c] = (__bf16)acc[i][j][r];
      }
    }
  }
}

// ---------------------------------------------------------------------------
// Kernel 2: global_q[b,d] = sum_n q[b,n,d] * softmax_d(q*alpha*scale)
// One wave per 64 rows; 24 d-elements per lane; wave32 shfl reductions.
// ---------------------------------------------------------------------------
__global__ __launch_bounds__(256) void reduce_alpha(const float* __restrict__ q,
                                                    const float* __restrict__ alpha,
                                                    float* __restrict__ gq) {
  int lane = threadIdx.x & 31;
  int wave = threadIdx.x >> 5;
  int gw = blockIdx.x * 8 + wave;      // 0..511
  int b  = gw >> 6;                    // 64 waves per batch
  int ws = gw & 63;
  float al[24], acc[24];
#pragma unroll
  for (int j = 0; j < 24; ++j) { al[j] = alpha[lane + 32 * j] * SCALE; acc[j] = 0.0f; }
  for (int rr = 0; rr < 64; ++rr) {
    int row = b * 4096 + ws * 64 + rr;
    const float* qr = q + (size_t)row * DIMK;
    float qv[24], tv[24];
    float m = -3.4e38f;
#pragma unroll
    for (int j = 0; j < 24; ++j) {
      float v = qr[lane + 32 * j];
      qv[j] = v; float t = v * al[j]; tv[j] = t; m = fmaxf(m, t);
    }
#pragma unroll
    for (int off = 16; off > 0; off >>= 1) m = fmaxf(m, __shfl_xor(m, off, 32));
    float s = 0.0f;
#pragma unroll
    for (int j = 0; j < 24; ++j) { float e = __expf(tv[j] - m); tv[j] = e; s += e; }
#pragma unroll
    for (int off = 16; off > 0; off >>= 1) s += __shfl_xor(s, off, 32);
    float inv = 1.0f / s;
#pragma unroll
    for (int j = 0; j < 24; ++j) acc[j] += qv[j] * tv[j] * inv;
  }
#pragma unroll
  for (int j = 0; j < 24; ++j) atomicAdd(&gq[b * DIMK + lane + 32 * j], acc[j]);
}

// ---------------------------------------------------------------------------
// Kernel 3: p = global_q (.) k ;  global_k[b,d] = sum_n p * softmax_d(p*beta*scale)
// ---------------------------------------------------------------------------
__global__ __launch_bounds__(256) void reduce_beta(const __bf16* __restrict__ kb,
                                                   const float* __restrict__ beta,
                                                   const float* __restrict__ gq,
                                                   float* __restrict__ gk) {
  int lane = threadIdx.x & 31;
  int wave = threadIdx.x >> 5;
  int gw = blockIdx.x * 8 + wave;
  int b  = gw >> 6;
  int ws = gw & 63;
  float be[24], gql[24], acc[24];
#pragma unroll
  for (int j = 0; j < 24; ++j) {
    be[j]  = beta[lane + 32 * j] * SCALE;
    gql[j] = gq[b * DIMK + lane + 32 * j];
    acc[j] = 0.0f;
  }
  for (int rr = 0; rr < 64; ++rr) {
    int row = b * 4096 + ws * 64 + rr;
    const __bf16* kr = kb + (size_t)row * DIMK;
    float pv[24], tv[24];
    float m = -3.4e38f;
#pragma unroll
    for (int j = 0; j < 24; ++j) {
      float p = gql[j] * (float)kr[lane + 32 * j];
      pv[j] = p; float t = p * be[j]; tv[j] = t; m = fmaxf(m, t);
    }
#pragma unroll
    for (int off = 16; off > 0; off >>= 1) m = fmaxf(m, __shfl_xor(m, off, 32));
    float s = 0.0f;
#pragma unroll
    for (int j = 0; j < 24; ++j) { float e = __expf(tv[j] - m); tv[j] = e; s += e; }
#pragma unroll
    for (int off = 16; off > 0; off >>= 1) s += __shfl_xor(s, off, 32);
    float inv = 1.0f / s;
#pragma unroll
    for (int j = 0; j < 24; ++j) acc[j] += pv[j] * tv[j] * inv;
  }
#pragma unroll
  for (int j = 0; j < 24; ++j) atomicAdd(&gk[b * DIMK + lane + 32 * j], acc[j]);
}

// ---------------------------------------------------------------------------
// Kernel 4: out = (global_k (.) v) @ Wr + q   (K-axis scaling applied during
// LDS staging; same double-buffered WMMA tiling as gemm_qkv).
// ---------------------------------------------------------------------------
__global__ __launch_bounds__(256) void gemm_out(const __bf16* __restrict__ vb,
                                                const float* __restrict__ gk,
                                                const __bf16* __restrict__ Bp,
                                                const float* __restrict__ q,
                                                float* __restrict__ out) {
  __shared__ __bf16 lA[2 * LA_BUF];
  int tid  = threadIdx.x;
  int lane = tid & 31;
  int wave = tid >> 5;
  int wm   = wave & 3;
  int wn   = wave >> 2;
  int mbase = blockIdx.x * 128;
  int cbase = blockIdx.y * 128;        // 0..767 (within Wr)
  int bb    = mbase >> 12;             // 4096 rows per batch; tile never straddles

  v8f acc[2][4];
#pragma unroll
  for (int i = 0; i < 2; ++i)
#pragma unroll
    for (int j = 0; j < 4; ++j)
#pragma unroll
      for (int r = 0; r < 8; ++r) acc[i][j][r] = 0.0f;

  int sm    = mbase + wave * 16 + (lane & 15);
  int skoff = (lane >> 4) * 8;
  const __bf16* srow = vb + (size_t)sm * DIMK + skoff;
  const float*  gkb  = gk + bb * DIMK + skoff;

  auto stage = [&](int kt, int buf) {
    const __bf16* s = srow + kt * 32;
    const float*  g = gkb + kt * 32;
    v8bf h0 = *(const v8bf*)(s);        // K = base..base+7
    v8bf h1 = *(const v8bf*)(s + 16);   // K = base+16..base+23
    __bf16* d = &lA[buf * LA_BUF + tid * 16];
#pragma unroll
    for (int e = 0; e < 8; ++e) d[e]     = (__bf16)((float)h0[e] * g[e]);
#pragma unroll
    for (int e = 0; e < 8; ++e) d[8 + e] = (__bf16)((float)h1[e] * g[16 + e]);
  };

  stage(0, 0);
  for (int kt = 0; kt < KT_STEPS; ++kt) {
    __syncthreads();
    int cur = kt & 1;
    if (kt + 1 < KT_STEPS) {
      __builtin_prefetch(srow + (kt + 2) * 32, 0, 3);
      stage(kt + 1, cur ^ 1);
    }
    v16bf a0 = *(const v16bf*)&lA[cur * LA_BUF + ((2 * wm + 0) * 32 + lane) * 16];
    v16bf a1 = *(const v16bf*)&lA[cur * LA_BUF + ((2 * wm + 1) * 32 + lane) * 16];
#pragma unroll
    for (int j = 0; j < 4; ++j) {
      int ct = 144 + (cbase >> 4) + wn * 4 + j;   // Wr segment of packed B
      v16bf bfrag = *(const v16bf*)(Bp + (((size_t)ct * KT_STEPS + kt) * 32 + lane) * 16);
      acc[0][j] = __builtin_amdgcn_wmma_f32_16x16x32_bf16(false, a0, false, bfrag,
                                                          (short)0, acc[0][j], false, false);
      acc[1][j] = __builtin_amdgcn_wmma_f32_16x16x32_bf16(false, a1, false, bfrag,
                                                          (short)0, acc[1][j], false, false);
    }
  }

#pragma unroll
  for (int i = 0; i < 2; ++i) {
    int rb = mbase + (2 * wm + i) * 16 + ((lane < 16) ? 0 : 8);
#pragma unroll
    for (int j = 0; j < 4; ++j) {
      int c = cbase + (wn * 4 + j) * 16 + (lane & 15);
#pragma unroll
      for (int r = 0; r < 8; ++r) {
        size_t idx = (size_t)(rb + r) * DIMK + c;
        out[idx] = acc[i][j][r] + q[idx];
      }
    }
  }
}

// ---------------------------------------------------------------------------
// Host launcher
// ---------------------------------------------------------------------------
extern "C" void kernel_launch(void* const* d_in, const int* in_sizes, int n_in,
                              void* d_out, int out_size, void* d_ws, size_t ws_size,
                              hipStream_t stream) {
  (void)in_sizes; (void)n_in; (void)out_size; (void)ws_size;
  const float* x     = (const float*)d_in[0];
  const float* Wq    = (const float*)d_in[1];
  const float* Wk    = (const float*)d_in[2];
  const float* Wv    = (const float*)d_in[3];
  const float* Wr    = (const float*)d_in[4];
  const float* alpha = (const float*)d_in[5];
  const float* beta  = (const float*)d_in[6];
  float* out = (float*)d_out;

  // workspace layout (bytes); total ~197 MB
  char* ws = (char*)d_ws;
  float*  q  = (float*)(ws + 0);                       // 32768*768*4  = 100663296
  __bf16* kb = (__bf16*)(ws + 100663296);              // 32768*768*2  =  50331648
  __bf16* vb = (__bf16*)(ws + 150994944);              // 32768*768*2  =  50331648
  __bf16* Bp = (__bf16*)(ws + 201326592);              // 192*24*32*16*2 = 4718592
  float*  gq = (float*)(ws + 206045184);               // 8*768*4 = 24576
  float*  gk = (float*)(ws + 206069760);               // 8*768*4 = 24576

  // 0) pack all four weight matrices into bf16 WMMA B-fragments
  pack_weights<<<dim3((CT_TOTAL * KT_STEPS * 32) / 256), 256, 0, stream>>>(Wq, Wk, Wv, Wr, Bp);
  // 1) zero both global accumulators (gq,gk are contiguous: 12288 floats)
  zero_f32<<<dim3(48), 256, 0, stream>>>(gq, 2 * 8 * DIMK);
  // 2) fused q/k/v GEMM (col space 3*768 = 18 blocks of 128)
  gemm_qkv<<<dim3(NROWS / 128, 18), 256, 0, stream>>>(x, Bp, q, kb, vb);
  // 3) softmax-weighted reduction -> global_q
  reduce_alpha<<<dim3(64), 256, 0, stream>>>(q, alpha, gq);
  // 4) softmax-weighted reduction -> global_k
  reduce_beta<<<dim3(64), 256, 0, stream>>>(kb, beta, gq, gk);
  // 5) output GEMM with K-axis scaling + residual q
  gemm_out<<<dim3(NROWS / 128, DIMK / 128), 256, 0, stream>>>(vb, gk, Bp, q, out);
}
